// RecurrentCorrector_1125281432254
// MI455X (gfx1250) — matrix operand order repaired and checked by reading, not compiled, not run on hardware
//
#include <hip/hip_runtime.h>
#include <hip/hip_bf16.h>

// ---------------------------------------------------------------------------
// RecurrentCorrector for MI455X (gfx1250, wave32, WMMA bf16 16x16x32)
// B=32 T=2048 E=H=D=512, G=3H=1536
// ---------------------------------------------------------------------------

typedef __bf16 bf16_t;
typedef __attribute__((ext_vector_type(16))) __bf16 v16bf;
typedef __attribute__((ext_vector_type(8)))  __bf16 v8bf;
typedef __attribute__((ext_vector_type(8)))  float  v8f;

#define WMMA_BF16(a, b, c) \
  __builtin_amdgcn_wmma_f32_16x16x32_bf16(false, (a), false, (b), (short)0, (c), false, false)

__device__ __forceinline__ float sigmoidf_(float x) { return 1.f / (1.f + __expf(-x)); }

// Defeat LICM without destroying pointer provenance: add an opaque zero
// offset (re-materialized in an SGPR every iteration). The GEP stays rooted
// at the kernarg pointer, so InferAddressSpaces keeps the loads on the
// GLOBAL path (global_load_b128, LOADcnt only) instead of FLAT.
__device__ __forceinline__ const bf16_t* launder(const bf16_t* p) {
  unsigned int zero = 0;
  asm volatile("" : "+s"(zero));
  return p + zero;
}

// A-fragment (16x32 bf16, MxK) per CDNA5 ISA layout:
//   lane L: row M = L%16.  halves 0..7 : K = kbase + (L/16)*8 + i
//                          halves 8..15: K = kbase + 16 + (L/16)*8 + (i-8)
__device__ __forceinline__ v16bf make_a(const bf16_t* rowp, int kbase, int lane) {
  const int ksub = ((lane >> 4) << 3);
  v8bf lo = *(const v8bf*)(rowp + kbase + ksub);
  v8bf hi = *(const v8bf*)(rowp + kbase + 16 + ksub);
  return __builtin_shufflevector(lo, hi, 0,1,2,3,4,5,6,7,8,9,10,11,12,13,14,15);
}

// B-fragment (32x16 bf16, KxN), pre-packed so lane L reads 16 contiguous halfs.
__device__ __forceinline__ v16bf load_bfrag(const bf16_t* __restrict__ Bp, int frag, int lane) {
  return *(const v16bf*)(Bp + ((size_t)frag << 9) + (lane << 4));
}

// ---------------------------------------------------------------------------
// f32 (B,T,E) -> bf16 (T*B, E)  (swapaxes during conversion)
// ---------------------------------------------------------------------------
__global__ void cvt_bt_kernel(const float* __restrict__ X, bf16_t* __restrict__ out,
                              int Bc, int T, int E) {
  size_t idx = (size_t)blockIdx.x * blockDim.x + threadIdx.x;
  size_t total = (size_t)Bc * T * E;
  if (idx >= total) return;
  size_t e = idx % E;
  size_t tb = idx / E;
  size_t b = tb % Bc;
  size_t t = tb / Bc;
  out[idx] = (bf16_t)X[(b * T + t) * E + e];
}

// ---------------------------------------------------------------------------
// Pack W (srcRows x K slice of a row-major f32 matrix, col offset) into
// WMMA B fragments: frag (nt,kt) -> 512 halfs, out[frag*512 + lane*16 + i] =
//   W[nt*16 + lane%16][kt*32 + (lane/16)*8 + (i%8) + (i/8)*16]   (rows >= srcRows -> 0)
// ---------------------------------------------------------------------------
__global__ void pack_b_kernel(const float* __restrict__ W, int srcRows, int srcStride,
                              int colOff, int G, int K, bf16_t* __restrict__ out) {
  int idx = blockIdx.x * blockDim.x + threadIdx.x;
  int total = G * K;
  if (idx >= total) return;
  int f = idx >> 9;
  int r = idx & 511;
  int l = r >> 4, i = r & 15;
  int KT = K >> 5;
  int nt = f / KT, kt = f % KT;
  int row = nt * 16 + (l & 15);
  int col = kt * 32 + ((l >> 4) << 3) + (i & 7) + ((i >> 3) << 4);
  float v = (row < srcRows) ? W[(size_t)row * srcStride + colOff + col] : 0.f;
  out[idx] = (bf16_t)v;
}

// ---------------------------------------------------------------------------
// Big GEMM: Out[M,N] = A[M,K](bf16 row-major) * Bpacked + bias, all-wave WMMA.
// Block = 8 waves; wave -> one 16-row M tile x 8 N tiles.
// ---------------------------------------------------------------------------
template <bool OUT_BF16, bool ACCUM>
__global__ void __launch_bounds__(256) gemm_wmma(const bf16_t* __restrict__ A,
                                                 const bf16_t* __restrict__ Bp,
                                                 const float* __restrict__ bias,
                                                 void* __restrict__ Out,
                                                 int M, int K, int N) {
  const int lane = threadIdx.x & 31;
  const int wave = threadIdx.x >> 5;
  const int mt = blockIdx.y * 8 + wave;
  const int nt0 = blockIdx.x * 8;
  const int KT = K >> 5;
  if (mt * 16 >= M) return;
  const bf16_t* arow = A + (size_t)(mt * 16 + (lane & 15)) * K;

  v8f acc[8] = {};
  for (int kt = 0; kt < KT; ++kt) {
    v16bf a = make_a(arow, kt * 32, lane);
#pragma unroll
    for (int n = 0; n < 8; ++n) {
      v16bf b = load_bfrag(Bp, (nt0 + n) * KT + kt, lane);
      acc[n] = WMMA_BF16(a, b, acc[n]);
    }
  }
  const int mloc = (lane >> 4) << 3;
  const int ncol0 = nt0 * 16 + (lane & 15);
#pragma unroll
  for (int n = 0; n < 8; ++n) {
    int col = ncol0 + n * 16;
    float bv = bias ? bias[col] : 0.f;
#pragma unroll
    for (int g = 0; g < 8; ++g) {
      int m = mt * 16 + g + mloc;
      float v = acc[n][g] + bv;
      size_t idx = (size_t)m * N + col;
      if (OUT_BF16) {
        ((bf16_t*)Out)[idx] = (bf16_t)v;
      } else {
        float* o = (float*)Out;
        if (ACCUM) o[idx] += v; else o[idx] = v;
      }
    }
  }
}

// ---------------------------------------------------------------------------
// Scan 1: GRU over T steps. 2 blocks (batch halves, M=16), 512 thr = 16 waves.
// Wave w owns h-column tiles {w, w+16}; per tile computes gate tiles
// {ht, ht+32, ht+64} of h @ W_hh1^T via WMMA over K=512.
// h kept in LDS: bf16 (WMMA A operand) + f32 (gate math).
// Weight fragments are streamed from (L2-resident) global each step; the
// base pointer gets an opaque per-iteration zero offset so LICM cannot
// hoist+spill them while the loads stay on the GLOBAL path.
// ---------------------------------------------------------------------------
__global__ void __launch_bounds__(512, 1) scan1_kernel(const bf16_t* __restrict__ gi1,
                                                       const bf16_t* __restrict__ Whh1p,
                                                       const float* __restrict__ b_hh1,
                                                       bf16_t* __restrict__ Rbf) {
  constexpr int H = 512, G = 1536, T = 2048, Bc = 32, KT = 16;
  __shared__ bf16_t sh_h[16 * H];
  __shared__ float  sf_h[16 * H];
  const int lane = threadIdx.x & 31;
  const int wave = threadIdx.x >> 5;          // 0..15
  const int bbase = blockIdx.x * 16;
  const int mloc = (lane >> 4) << 3;

  for (int i = threadIdx.x; i < 16 * H; i += blockDim.x) { sh_h[i] = (bf16_t)0.f; sf_h[i] = 0.f; }
  __syncthreads();
  const bf16_t* hrow = sh_h + (lane & 15) * H;

  float bh[2][3];
#pragma unroll
  for (int p = 0; p < 2; ++p) {
    int col = (wave + p * 16) * 16 + (lane & 15);
#pragma unroll
    for (int j = 0; j < 3; ++j) bh[p][j] = b_hh1[j * 512 + col];
  }

  for (int t = 0; t < T; ++t) {
    const bf16_t* Wp = launder(Whh1p);       // defeat LICM hoist+spill
    float hn[2][8];
    const size_t growbase = (size_t)t * Bc + bbase;
#pragma unroll
    for (int p = 0; p < 2; ++p) {
      const int ht = wave + p * 16;
      v8f acc[3] = {};
      for (int kt = 0; kt < KT; ++kt) {
        v16bf a = make_a(hrow, kt * 32, lane);
#pragma unroll
        for (int j = 0; j < 3; ++j) {
          v16bf b = load_bfrag(Wp, (j * 32 + ht) * KT + kt, lane);
          acc[j] = WMMA_BF16(a, b, acc[j]);
        }
      }
      const int col = ht * 16 + (lane & 15);
#pragma unroll
      for (int g = 0; g < 8; ++g) {
        int m = g + mloc;
        const bf16_t* gir = gi1 + (growbase + m) * (size_t)G;
        float r  = sigmoidf_((float)gir[col]        + acc[0][g] + bh[p][0]);
        float z  = sigmoidf_((float)gir[512 + col]  + acc[1][g] + bh[p][1]);
        float nn = tanhf(    (float)gir[1024 + col] + r * (acc[2][g] + bh[p][2]));
        hn[p][g] = (1.f - z) * nn + z * sf_h[m * H + col];
      }
    }
    __syncthreads();                         // all reads of h_prev complete
#pragma unroll
    for (int p = 0; p < 2; ++p) {
      const int col = (wave + p * 16) * 16 + (lane & 15);
#pragma unroll
      for (int g = 0; g < 8; ++g) {
        int m = g + mloc;
        float v = hn[p][g];
        bf16_t vb = (bf16_t)v;
        sf_h[m * H + col] = v;
        sh_h[m * H + col] = vb;
        Rbf[(growbase + m) * (size_t)H + col] = vb;
      }
    }
    __syncthreads();
  }
}

// ---------------------------------------------------------------------------
// Scan 2: GRU + corrector head. Same decomposition; per step two K=512 WMMA
// GEMMs (h@W_hh2^T, prev@W_prev^T, kept separate because only the hh part is
// scaled by r inside the n-gate), then corr = tanh(h@W_corr^T)·(1-rew),
// rew = sigmoid(h@W_rew^T) via a padded N=16 WMMA tile on wave 0.
// ---------------------------------------------------------------------------
__global__ void __launch_bounds__(512, 1) scan2_kernel(const float* __restrict__ gi2,
                                                       const bf16_t* __restrict__ Whh2p,
                                                       const bf16_t* __restrict__ Wprevp,
                                                       const bf16_t* __restrict__ Wcorrp,
                                                       const bf16_t* __restrict__ Wrewp,
                                                       const float* __restrict__ b_hh2,
                                                       const float* __restrict__ b_corr,
                                                       const float* __restrict__ b_rew,
                                                       float* __restrict__ out_corr,
                                                       float* __restrict__ out_rew) {
  constexpr int H = 512, D = 512, G = 1536, T = 2048, Bc = 32, KT = 16;
  __shared__ bf16_t sh_h[16 * H];
  __shared__ float  sf_h[16 * H];
  __shared__ bf16_t sh_p[16 * D];
  __shared__ float  s_rew[16];
  const int lane = threadIdx.x & 31;
  const int wave = threadIdx.x >> 5;
  const int bbase = blockIdx.x * 16;
  const int mloc = (lane >> 4) << 3;

  for (int i = threadIdx.x; i < 16 * H; i += blockDim.x) {
    sh_h[i] = (bf16_t)0.f; sf_h[i] = 0.f; sh_p[i] = (bf16_t)0.f;
  }
  __syncthreads();
  const bf16_t* hrow = sh_h + (lane & 15) * H;
  const bf16_t* prow = sh_p + (lane & 15) * D;

  float bh[2][3];
#pragma unroll
  for (int p = 0; p < 2; ++p) {
    int col = (wave + p * 16) * 16 + (lane & 15);
#pragma unroll
    for (int j = 0; j < 3; ++j) bh[p][j] = b_hh2[j * 512 + col];
  }
  const float brew = b_rew[0];

  for (int t = 0; t < T; ++t) {
    const bf16_t* Whh = launder(Whh2p);      // defeat LICM hoist+spill
    const bf16_t* Wpv = launder(Wprevp);
    float hn[2][8];
    const size_t growbase = (size_t)t * Bc + bbase;
#pragma unroll
    for (int p = 0; p < 2; ++p) {
      const int ht = wave + p * 16;
      v8f ah[3] = {}, ap[3] = {};
      for (int kt = 0; kt < KT; ++kt) {
        v16bf a1 = make_a(hrow, kt * 32, lane);
        v16bf a2 = make_a(prow, kt * 32, lane);
#pragma unroll
        for (int j = 0; j < 3; ++j) {
          ah[j] = WMMA_BF16(a1, load_bfrag(Whh, (j * 32 + ht) * KT + kt, lane), ah[j]);
          ap[j] = WMMA_BF16(a2, load_bfrag(Wpv, (j * 32 + ht) * KT + kt, lane), ap[j]);
        }
      }
      const int col = ht * 16 + (lane & 15);
#pragma unroll
      for (int g = 0; g < 8; ++g) {
        int m = g + mloc;
        const float* gir = gi2 + (growbase + m) * (size_t)G;
        float ir  = gir[col]        + ap[0][g];
        float iz  = gir[512 + col]  + ap[1][g];
        float inn = gir[1024 + col] + ap[2][g];
        float r  = sigmoidf_(ir + ah[0][g] + bh[p][0]);
        float z  = sigmoidf_(iz + ah[1][g] + bh[p][1]);
        float nn = tanhf(inn + r * (ah[2][g] + bh[p][2]));
        hn[p][g] = (1.f - z) * nn + z * sf_h[m * H + col];
      }
    }
    __syncthreads();
#pragma unroll
    for (int p = 0; p < 2; ++p) {
      const int col = (wave + p * 16) * 16 + (lane & 15);
#pragma unroll
      for (int g = 0; g < 8; ++g) {
        int m = g + mloc;
        sf_h[m * H + col] = hn[p][g];
        sh_h[m * H + col] = (bf16_t)hn[p][g];
      }
    }
    __syncthreads();                          // h_new visible in LDS

    // ----- corrector / reward head on h_new -----
    const bf16_t* Wco = launder(Wcorrp);
    const bf16_t* Wrw = launder(Wrewp);
    v8f ca[2] = {};
#pragma unroll
    for (int p = 0; p < 2; ++p) {
      const int ct = wave + p * 16;
      for (int kt = 0; kt < KT; ++kt) {
        v16bf a = make_a(hrow, kt * 32, lane);
        ca[p] = WMMA_BF16(a, load_bfrag(Wco, ct * KT + kt, lane), ca[p]);
      }
    }
    if (wave == 0) {
      v8f ra = {};
      for (int kt = 0; kt < KT; ++kt) {
        v16bf a = make_a(hrow, kt * 32, lane);
        ra = WMMA_BF16(a, load_bfrag(Wrw, kt, lane), ra);
      }
      if ((lane & 15) == 0) {                 // column 0 of padded rew tile
#pragma unroll
        for (int g = 0; g < 8; ++g) s_rew[g + mloc] = sigmoidf_(ra[g] + brew);
      }
    }
    __syncthreads();
#pragma unroll
    for (int p = 0; p < 2; ++p) {
      const int col = (wave + p * 16) * 16 + (lane & 15);
#pragma unroll
      for (int g = 0; g < 8; ++g) {
        int m = g + mloc;
        float rw = s_rew[m];
        float c = tanhf(ca[p][g] + b_corr[col]) * (1.f - rw);
        sh_p[m * D + col] = (bf16_t)c;        // prev for next step
        out_corr[((size_t)(bbase + m) * T + t) * D + col] = c;
      }
    }
    if (wave == 0 && (lane & 15) == 0) {
#pragma unroll
      for (int g = 0; g < 8; ++g) {
        int m = g + mloc;
        out_rew[(size_t)(bbase + m) * T + t] = s_rew[m];
      }
    }
    __syncthreads();
  }
}

// ---------------------------------------------------------------------------
extern "C" void kernel_launch(void* const* d_in, const int* in_sizes, int n_in,
                              void* d_out, int out_size, void* d_ws, size_t ws_size,
                              hipStream_t stream) {
  (void)in_sizes; (void)n_in; (void)out_size; (void)ws_size;
  constexpr int Bc = 32, T = 2048, E = 512, H = 512, D = 512, G = 1536;
  constexpr int TB = Bc * T;                  // 65536 rows

  const float* rin    = (const float*)d_in[0];
  const float* ystat  = (const float*)d_in[3];
  const float* W_ih1  = (const float*)d_in[4];
  const float* W_hh1  = (const float*)d_in[5];
  const float* b_ih1  = (const float*)d_in[6];
  const float* b_hh1  = (const float*)d_in[7];
  const float* W_ih2  = (const float*)d_in[8];
  const float* W_hh2  = (const float*)d_in[9];
  const float* b_ih2  = (const float*)d_in[10];
  const float* b_hh2  = (const float*)d_in[11];
  const float* W_corr = (const float*)d_in[12];
  const float* b_corr = (const float*)d_in[13];
  const float* W_rew  = (const float*)d_in[14];
  const float* b_rew  = (const float*)d_in[15];

  char* ws = (char*)d_ws;
  size_t off = 0;
  auto take = [&](size_t bytes) -> char* {
    off = (off + 255) & ~(size_t)255;
    char* p = ws + off;
    off += bytes;
    return p;
  };

  bf16_t* Xbf    = (bf16_t*)take((size_t)TB * E * 2);
  bf16_t* Ybf    = (bf16_t*)take((size_t)TB * D * 2);
  bf16_t* Rbf    = (bf16_t*)take((size_t)TB * H * 2);
  bf16_t* gi1    = (bf16_t*)take((size_t)TB * G * 2);
  float*  gi2    = (float*) take((size_t)TB * G * 4);
  bf16_t* Wih1p  = (bf16_t*)take((size_t)G * E * 2);
  bf16_t* Whh1p  = (bf16_t*)take((size_t)G * H * 2);
  bf16_t* Wyp    = (bf16_t*)take((size_t)G * D * 2);
  bf16_t* Wrp    = (bf16_t*)take((size_t)G * H * 2);
  bf16_t* Wprevp = (bf16_t*)take((size_t)G * D * 2);
  bf16_t* Whh2p  = (bf16_t*)take((size_t)G * H * 2);
  bf16_t* Wcorrp = (bf16_t*)take((size_t)D * H * 2);
  bf16_t* Wrewp  = (bf16_t*)take((size_t)16 * H * 2);

  // 1) convert + (B,T)->(T,B) transpose of activations, f32->bf16
  {
    size_t tot = (size_t)TB * E;
    dim3 gridc((unsigned)((tot + 255) / 256));
    cvt_bt_kernel<<<gridc, 256, 0, stream>>>(rin,   Xbf, Bc, T, E);
    cvt_bt_kernel<<<gridc, 256, 0, stream>>>(ystat, Ybf, Bc, T, D);
  }

  // 2) pack all weight matrices into WMMA B-fragment layout
  auto pack = [&](const float* W, int srcRows, int stride, int colOff, int Gp, int K, bf16_t* out) {
    int tot = Gp * K;
    pack_b_kernel<<<dim3((tot + 255) / 256), 256, 0, stream>>>(W, srcRows, stride, colOff, Gp, K, out);
  };
  pack(W_ih1, G, E, 0,      G, E, Wih1p);
  pack(W_hh1, G, H, 0,      G, H, Whh1p);
  pack(W_ih2, G, 1536, 0,    G, D, Wyp);     // W_ih2[:, 0:512]    (y_states part)
  pack(W_ih2, G, 1536, 512,  G, H, Wrp);     // W_ih2[:, 512:1024] (rnn_out part)
  pack(W_ih2, G, 1536, 1024, G, D, Wprevp);  // W_ih2[:, 1024:1536] (prev corr)
  pack(W_hh2, G, H, 0,      G, H, Whh2p);
  pack(W_corr, D, H, 0,     D, H, Wcorrp);
  pack(W_rew,  1, H, 0,    16, H, Wrewp);    // pad to one N tile, rows>=1 zeroed

  // 3) gi1 = X @ W_ih1^T + b_ih1   (bf16 out)
  gemm_wmma<true,  false><<<dim3(G / 128, TB / 128), 256, 0, stream>>>(Xbf, Wih1p, b_ih1, gi1, TB, E, G);
  // 4) gi2 = y @ W_y^T + b_ih2     (f32 out)
  gemm_wmma<false, false><<<dim3(G / 128, TB / 128), 256, 0, stream>>>(Ybf, Wyp,   b_ih2, gi2, TB, D, G);
  // 5) sequential GRU scan #1 (2 independent batch halves)
  scan1_kernel<<<dim3(2), 512, 0, stream>>>(gi1, Whh1p, b_hh1, Rbf);
  // 6) gi2 += rnn_out @ W_r^T
  gemm_wmma<false, true><<<dim3(G / 128, TB / 128), 256, 0, stream>>>(Rbf, Wrp, nullptr, gi2, TB, H, G);
  // 7) sequential GRU scan #2 + corrector head, writes outputs (B,T,*) f32
  float* out_corr = (float*)d_out;
  float* out_rew  = out_corr + (size_t)Bc * T * D;
  scan2_kernel<<<dim3(2), 512, 0, stream>>>(gi2, Whh2p, Wprevp, Wcorrp, Wrewp,
                                            b_hh2, b_corr, b_rew, out_corr, out_rew);
}
